// MoERunner_12412455485619
// MI455X (gfx1250) — matrix operand order repaired and checked by reading, not compiled, not run on hardware
//
#include <hip/hip_runtime.h>

// ---------------- problem constants ----------------
#define T_TOK 16384
#define H_DIM 1024
#define I_DIM 2048
#define E_EXP 8
#define K_TOP 2
#define N_ASSIGN (T_TOK * K_TOP)                  // 32768
#define TILE_M 64
#define MAX_TILES ((N_ASSIGN / TILE_M) + E_EXP)   // 520 worst case (64-aligned pads)
#define PAD_MAX (MAX_TILES * TILE_M)              // 33280 padded assignment slots

typedef __attribute__((ext_vector_type(16))) __bf16 v16bf;
typedef __attribute__((ext_vector_type(8)))  __bf16 v8bf;
typedef __attribute__((ext_vector_type(8)))  float  v8f;
typedef __attribute__((ext_vector_type(4)))  int    i4;

// -------- gfx1250 async load-to-LDS (ASYNCcnt) with safe fallback --------
#ifdef __has_builtin
#if __has_builtin(__builtin_amdgcn_global_load_async_to_lds_b128)
#define HAVE_ASYNC_LDS 1
#endif
#if __has_builtin(__builtin_amdgcn_s_wait_asynccnt)
#define HAVE_WAIT_ASYNC 1
#endif
#endif

__device__ __forceinline__ void cp16(void* lds_dst, const void* gsrc) {
#ifdef HAVE_ASYNC_LDS
    __builtin_amdgcn_global_load_async_to_lds_b128(
        (__attribute__((address_space(1))) i4*)gsrc,
        (__attribute__((address_space(3))) i4*)lds_dst, 0, 0);
#else
    *(float4*)lds_dst = *(const float4*)gsrc;
#endif
}

__device__ __forceinline__ void stage_join() {
#ifdef HAVE_WAIT_ASYNC
    __builtin_amdgcn_s_wait_asynccnt(0);
#endif
    __syncthreads();
}

__device__ __forceinline__ v8f wmma_bf16(v16bf a, v16bf b, v8f c) {
    return __builtin_amdgcn_wmma_f32_16x16x32_bf16(false, a, false, b,
                                                   (short)0, c, false, false);
}

// silu via v_exp_f32 + v_rcp_f32 (no IEEE division chain)
__device__ __forceinline__ float silu_fast(float g) {
    return g * __builtin_amdgcn_rcpf(1.0f + __expf(-g));
}

// ============ routing ============
__global__ void moe_init(int* counts, int* cursors, int* tile_expert,
                         int* assign_token, float* assign_w) {
    int i = blockIdx.x * blockDim.x + threadIdx.x;
    if (i < PAD_MAX) { assign_token[i] = -1; assign_w[i] = 0.0f; }
    if (i < MAX_TILES) tile_expert[i] = -1;
    if (i < E_EXP) { counts[i] = 0; cursors[i] = 0; }
}

__global__ void moe_hist(const int* __restrict__ topk_ids, int* counts) {
    int i = blockIdx.x * blockDim.x + threadIdx.x;
    if (i < N_ASSIGN) atomicAdd(&counts[topk_ids[i]], 1);
}

__global__ void moe_scan(const int* __restrict__ counts, int* offs, int* tile_expert) {
    if (blockIdx.x == 0 && threadIdx.x == 0) {
        int off = 0;
        for (int e = 0; e < E_EXP; ++e) {
            offs[e] = off;
            int c = (counts[e] + TILE_M - 1) & ~(TILE_M - 1);
            for (int t = off / TILE_M; t < (off + c) / TILE_M; ++t) tile_expert[t] = e;
            off += c;
        }
        offs[E_EXP] = off;
    }
}

__global__ void moe_scatter(const int* __restrict__ topk_ids,
                            const float* __restrict__ topk_w,
                            const int* __restrict__ offs, int* cursors,
                            int* assign_token, float* assign_w) {
    int i = blockIdx.x * blockDim.x + threadIdx.x;
    if (i >= N_ASSIGN) return;
    int e = topk_ids[i];
    int pos = offs[e] + atomicAdd(&cursors[e], 1);
    assign_token[pos] = i / K_TOP;
    assign_w[pos] = topk_w[i];
}

__global__ void moe_zero_out(float* __restrict__ out) {
    size_t i = (size_t)(blockIdx.x * blockDim.x + threadIdx.x) * 4;
    *(float4*)(out + i) = make_float4(0.f, 0.f, 0.f, 0.f);
}

// one-shot f32 -> bf16 conversion (streaming, 8 elems/thread)
__global__ void moe_cvt_bf16(const float* __restrict__ src, __bf16* __restrict__ dst,
                             size_t n) {
    size_t i = (size_t)(blockIdx.x * blockDim.x + threadIdx.x) * 8;
    if (i >= n) return;
    float4 a = *(const float4*)(src + i);
    float4 b = *(const float4*)(src + i + 4);
    union { __bf16 h[8]; float4 f; } u;
    u.h[0] = (__bf16)a.x; u.h[1] = (__bf16)a.y; u.h[2] = (__bf16)a.z; u.h[3] = (__bf16)a.w;
    u.h[4] = (__bf16)b.x; u.h[5] = (__bf16)b.y; u.h[6] = (__bf16)b.z; u.h[7] = (__bf16)b.w;
    *(float4*)(dst + i) = u.f;
}

// ============ stage 1: h = x @ W13^T, a = silu(h1)*h3 ============
// grid (MAX_TILES, I/128); 256 thr = 8 waves. WG tile: 64 tokens x 128 i (gate+up).
__global__ __launch_bounds__(256, 2)
void moe_stage1(const __bf16* __restrict__ xb, const __bf16* __restrict__ w13b,
                const int* __restrict__ tile_expert,
                const int* __restrict__ assign_token,
                __bf16* __restrict__ a_buf) {
    __shared__ int toks[TILE_M];
    __shared__ __align__(16) __bf16 ldsA[TILE_M * 32];   // [m][k]
    __shared__ __align__(16) __bf16 ldsB[256 * 32];      // [n][k]: n<128 gate, n>=128 up
    __shared__ __align__(16) __bf16 ldsO[TILE_M * 128];  // activations

    const int tile = blockIdx.x;
    const int e = tile_expert[tile];
    if (e < 0) return;
    const int iBase = blockIdx.y * 128;
    const int tid = threadIdx.x, lane = tid & 31, wave = tid >> 5;
    const int m = lane & 15, hi = lane >> 4;
    const int msub = wave >> 1, half = wave & 1;

    if (tid < TILE_M) toks[tid] = assign_token[tile * TILE_M + tid];
    __syncthreads();

    // A staging: 64 rows x 4 chunks(16B) = 256, one per thread (pad rows clamp to token 0)
    const int arow = tid >> 2, ach = tid & 3;
    int atok = toks[arow]; if (atok < 0) atok = 0;
    const __bf16* asrc = xb + (size_t)atok * H_DIM + ach * 8;
    __bf16* adst = &ldsA[arow * 32 + ach * 8];
    const __bf16* wbase = w13b + (size_t)e * (2 * I_DIM) * H_DIM;

    v8f acc1[4] = {{0}}, acc3[4] = {{0}};

    for (int k0 = 0; k0 < H_DIM; k0 += 32) {
        cp16(adst, asrc + k0);
        // B staging: 256 rows x 4 chunks = 1024, four per thread
#pragma unroll
        for (int c = 0; c < 4; ++c) {
            int cid = c * 256 + tid;
            int row = cid >> 2, ch = cid & 3;
            int ig = (row < 128) ? (iBase + row) : (I_DIM + iBase + (row - 128));
            cp16(&ldsB[row * 32 + ch * 8], wbase + (size_t)ig * H_DIM + k0 + ch * 8);
        }
        stage_join();

        union { v16bf v; v8bf h[2]; } A;
        A.h[0] = *(const v8bf*)(&ldsA[(msub * 16 + m) * 32 + hi * 16]);
        A.h[1] = *(const v8bf*)(&ldsA[(msub * 16 + m) * 32 + hi * 16 + 8]);
#pragma unroll
        for (int j = 0; j < 4; ++j) {
            int nsub = 4 * half + j;
            union { v16bf v; v8bf h[2]; } Bg, Bu;
            int rg = nsub * 16 + m;
            Bg.h[0] = *(const v8bf*)(&ldsB[rg * 32 + hi * 16]);
            Bg.h[1] = *(const v8bf*)(&ldsB[rg * 32 + hi * 16 + 8]);
            int ru = 128 + nsub * 16 + m;
            Bu.h[0] = *(const v8bf*)(&ldsB[ru * 32 + hi * 16]);
            Bu.h[1] = *(const v8bf*)(&ldsB[ru * 32 + hi * 16 + 8]);
            acc1[j] = wmma_bf16(A.v, Bg.v, acc1[j]);
            acc3[j] = wmma_bf16(A.v, Bu.v, acc3[j]);
        }
        __syncthreads();
    }

    // silu(gate)*up (gate/up accumulators share (lane,vgpr) mapping), LDS transpose out
#pragma unroll
    for (int j = 0; j < 4; ++j) {
        int nsub = 4 * half + j;
#pragma unroll
        for (int r = 0; r < 8; ++r) {
            float a = silu_fast(acc1[j][r]) * acc3[j][r];
            int row = msub * 16 + r + 8 * hi;
            ldsO[row * 128 + nsub * 16 + m] = (__bf16)a;
        }
    }
    __syncthreads();
    // store 64 x 128 bf16 tile: 1024 chunks, 4/thread
#pragma unroll
    for (int c = 0; c < 4; ++c) {
        int cid = c * 256 + tid;
        int row = cid >> 4, ch = cid & 15;
        *(float4*)(a_buf + ((size_t)tile * TILE_M + row) * I_DIM + iBase + ch * 8) =
            *(const float4*)(&ldsO[row * 128 + ch * 8]);
    }
}

// ============ stage 2: out += tw * (a @ W2^T) ============
// grid (MAX_TILES, H/128); 256 thr = 8 waves. WG tile: 64 tokens x 128 h-cols.
__global__ __launch_bounds__(256, 2)
void moe_stage2(const __bf16* __restrict__ a_buf, const __bf16* __restrict__ w2b,
                const int* __restrict__ tile_expert,
                const int* __restrict__ assign_token,
                const float* __restrict__ assign_w,
                float* __restrict__ out) {
    __shared__ int   toks[TILE_M];
    __shared__ float wts[TILE_M];
    __shared__ __align__(16) __bf16 ldsA[TILE_M * 32];
    __shared__ __align__(16) __bf16 ldsB[128 * 32];

    const int tile = blockIdx.x;
    const int e = tile_expert[tile];
    if (e < 0) return;
    const int hBase = blockIdx.y * 128;
    const int tid = threadIdx.x, lane = tid & 31, wave = tid >> 5;
    const int m = lane & 15, hi = lane >> 4;
    const int msub = wave >> 1, nhalf = wave & 1;

    if (tid < TILE_M) {
        toks[tid] = assign_token[tile * TILE_M + tid];
        wts[tid]  = assign_w[tile * TILE_M + tid];
    }
    __syncthreads();

    const __bf16* abase = a_buf + (size_t)tile * TILE_M * I_DIM;
    const __bf16* wbase = w2b + (size_t)e * H_DIM * I_DIM;

    v8f acc[4] = {{0}};

    for (int i0 = 0; i0 < I_DIM; i0 += 32) {
        // A: 64 rows x 4 chunks = 256, one per thread (bf16 already)
        {
            int row = tid >> 2, ch = tid & 3;
            cp16(&ldsA[row * 32 + ch * 8], abase + (size_t)row * I_DIM + i0 + ch * 8);
        }
        // B: 128 rows x 4 chunks = 512, two per thread
#pragma unroll
        for (int c = 0; c < 2; ++c) {
            int cid = c * 256 + tid;
            int row = cid >> 2, ch = cid & 3;
            cp16(&ldsB[row * 32 + ch * 8],
                 wbase + (size_t)(hBase + row) * I_DIM + i0 + ch * 8);
        }
        stage_join();

        union { v16bf v; v8bf h[2]; } A;
        A.h[0] = *(const v8bf*)(&ldsA[(msub * 16 + m) * 32 + hi * 16]);
        A.h[1] = *(const v8bf*)(&ldsA[(msub * 16 + m) * 32 + hi * 16 + 8]);
#pragma unroll
        for (int j = 0; j < 4; ++j) {
            int rB = (4 * nhalf + j) * 16 + m;
            union { v16bf v; v8bf h[2]; } B;
            B.h[0] = *(const v8bf*)(&ldsB[rB * 32 + hi * 16]);
            B.h[1] = *(const v8bf*)(&ldsB[rB * 32 + hi * 16 + 8]);
            acc[j] = wmma_bf16(A.v, B.v, acc[j]);
        }
        __syncthreads();
    }

    // scale by router weight, scatter-add (experts overlap per token)
#pragma unroll
    for (int j = 0; j < 4; ++j) {
        int col = hBase + (4 * nhalf + j) * 16 + m;
#pragma unroll
        for (int r = 0; r < 8; ++r) {
            int row = msub * 16 + r + 8 * hi;
            int tok = toks[row];
            if (tok >= 0)
                atomicAdd(out + (size_t)tok * H_DIM + col, wts[row] * acc[j][r]);
        }
    }
}

// ============ launcher ============
static inline size_t align_up(size_t v, size_t a) { return (v + a - 1) & ~(a - 1); }

extern "C" void kernel_launch(void* const* d_in, const int* in_sizes, int n_in,
                              void* d_out, int out_size, void* d_ws, size_t ws_size,
                              hipStream_t stream) {
    const float* x       = (const float*)d_in[0];
    const float* w13     = (const float*)d_in[1];
    const float* w2      = (const float*)d_in[2];
    const float* topk_w  = (const float*)d_in[3];
    const int*   topk_id = (const int*)d_in[4];
    float* out = (float*)d_out;

    char* p = (char*)d_ws;
    size_t off = 0;
    int*    counts       = (int*)(p + off);   off = align_up(off + E_EXP * 4, 256);
    int*    cursors      = (int*)(p + off);   off = align_up(off + E_EXP * 4, 256);
    int*    offs         = (int*)(p + off);   off = align_up(off + (E_EXP + 1) * 4, 256);
    int*    tile_expert  = (int*)(p + off);   off = align_up(off + MAX_TILES * 4, 256);
    int*    assign_token = (int*)(p + off);   off = align_up(off + (size_t)PAD_MAX * 4, 256);
    float*  assign_w     = (float*)(p + off); off = align_up(off + (size_t)PAD_MAX * 4, 256);
    __bf16* xb           = (__bf16*)(p + off); off = align_up(off + (size_t)T_TOK * H_DIM * 2, 256);
    __bf16* w13b         = (__bf16*)(p + off); off = align_up(off + (size_t)E_EXP * 2 * I_DIM * H_DIM * 2, 256);
    __bf16* w2b          = (__bf16*)(p + off); off = align_up(off + (size_t)E_EXP * H_DIM * I_DIM * 2, 256);
    __bf16* a_buf        = (__bf16*)(p + off); off += (size_t)PAD_MAX * I_DIM * 2;
    (void)ws_size; (void)in_sizes; (void)n_in; (void)out_size;

    // routing
    moe_init<<<(PAD_MAX + 255) / 256, 256, 0, stream>>>(counts, cursors, tile_expert,
                                                        assign_token, assign_w);
    moe_hist<<<N_ASSIGN / 256, 256, 0, stream>>>(topk_id, counts);
    moe_scan<<<1, 32, 0, stream>>>(counts, offs, tile_expert);
    moe_scatter<<<N_ASSIGN / 256, 256, 0, stream>>>(topk_id, topk_w, offs, cursors,
                                                    assign_token, assign_w);
    // one-shot bf16 conversion of activations + weights (pure HBM stream)
    moe_cvt_bf16<<<(T_TOK * H_DIM / 8 + 255) / 256, 256, 0, stream>>>(
        x, xb, (size_t)T_TOK * H_DIM);
    moe_cvt_bf16<<<((size_t)E_EXP * 2 * I_DIM * H_DIM / 8 + 255) / 256, 256, 0, stream>>>(
        w13, w13b, (size_t)E_EXP * 2 * I_DIM * H_DIM);
    moe_cvt_bf16<<<((size_t)E_EXP * H_DIM * I_DIM / 8 + 255) / 256, 256, 0, stream>>>(
        w2, w2b, (size_t)E_EXP * H_DIM * I_DIM);
    moe_zero_out<<<(T_TOK * H_DIM / 4) / 256, 256, 0, stream>>>(out);

    // fused expert GEMMs
    moe_stage1<<<dim3(MAX_TILES, I_DIM / 128), 256, 0, stream>>>(
        xb, w13b, tile_expert, assign_token, a_buf);
    moe_stage2<<<dim3(MAX_TILES, H_DIM / 128), 256, 0, stream>>>(
        a_buf, w2b, tile_expert, assign_token, assign_w, out);
}
